// QueryLocalConvAttentionDeep_62354335204048
// MI455X (gfx1250) — matrix operand order repaired and checked by reading, not compiled, not run on hardware
//
#include <hip/hip_runtime.h>
#include <hip/hip_bf16.h>
#include <math.h>

// ---------------------------------------------------------------------------
// QueryLocalConvAttentionDeep for MI455X (gfx1250, wave32, WMMA)
// All heavy GEMMs use v_wmma_f32_16x16x32_bf16. fp32 epilogues (bias, GELU,
// LayerNorm, softmax) preserve accuracy where it matters.
// ---------------------------------------------------------------------------

#define Bq   32
#define Nq   576
#define Gq   24
#define Dq   1024
#define HSq  4096
#define TDq  768
#define Hq   8
#define HDq  128
#define Tq   36          // 6x6 conv output positions
#define BTq  (Bq * Tq)   // 1152 query rows
#define ROWS (Bq * Nq)   // 18432 activation rows

typedef __attribute__((ext_vector_type(16))) __bf16 v16bf;
typedef __attribute__((ext_vector_type(8)))  float  v8f;

struct Frag32B { uint4 lo; uint4 hi; };

// ---- WMMA fragment loaders (CDNA5 16x16x32 bf16 layouts) -------------------

// A (M x K, row-major). lanes 0-15 row M=lane, K=0..15;
// lanes 16-31 row M=lane-16, K=16..31.
__device__ __forceinline__ v16bf load_a_frag(const __hip_bfloat16* __restrict__ A,
                                             int row, int kb, int K, int lane) {
    int k0 = kb + ((lane >> 4) << 4);                       // +0 or +16
    const uint4* p = (const uint4*)(A + (size_t)row * K + k0);
    Frag32B f;
    f.lo = p[0];     // K = k0   .. k0+7
    f.hi = p[1];     // K = k0+8 .. k0+15
    return __builtin_bit_cast(v16bf, f);
}

// B (K x N) supplied as W (N x K row-major, since C = A @ W^T).
// lanes 0-15: col N=lane, K=0..15 ; lanes 16-31: col N=lane-16, K=16..31.
__device__ __forceinline__ v16bf load_b_frag(const __hip_bfloat16* __restrict__ W,
                                             int ncol, int kb, int K, int lane) {
    int k0 = kb + ((lane >> 4) << 4);                       // +0 or +16
    const uint4* p = (const uint4*)(W + (size_t)ncol * K + k0);
    Frag32B f;
    f.lo = p[0];
    f.hi = p[1];
    return __builtin_bit_cast(v16bf, f);
}

__device__ __forceinline__ float gelu_exact(float x) {
    return 0.5f * x * (1.0f + erff(x * 0.7071067811865475f));
}

__device__ __forceinline__ void store_out(float* C, size_t i, float v) { C[i] = v; }
__device__ __forceinline__ void store_out(__hip_bfloat16* C, size_t i, float v) {
    C[i] = __float2bfloat16(v);
}

// ---- Blocked bf16 WMMA GEMM:  C(M,N) = act((A(M,K) @ W(N,K)^T + bias)*alpha)
// Block = 256 threads = 8 waves. Waves tiled 2(M) x 4(N); each wave owns a
// 64x32 C tile = 4x2 fragments. Requires M%128==0, N%128==0, K%32==0.
template <typename OutT, bool GELU>
__global__ __launch_bounds__(256, 2)
void gemm_bt_wmma(const __hip_bfloat16* __restrict__ A,
                  const __hip_bfloat16* __restrict__ W,
                  const float* __restrict__ bias,
                  OutT* __restrict__ C,
                  int M, int N, int K, float alpha) {
    const int lane = threadIdx.x & 31;
    const int wave = threadIdx.x >> 5;
    const int wm   = wave & 1;       // 2 waves along M
    const int wn   = wave >> 1;      // 4 waves along N
    const int mblk = blockIdx.y * 128 + wm * 64;
    const int nblk = blockIdx.x * 128 + wn * 32;

    v8f acc[4][2];
#pragma unroll
    for (int i = 0; i < 4; ++i)
#pragma unroll
        for (int j = 0; j < 2; ++j) acc[i][j] = (v8f){0.f,0.f,0.f,0.f,0.f,0.f,0.f,0.f};

    const int arow = lane & 15;
    for (int kb = 0; kb < K; kb += 32) {
        v16bf b0 = load_b_frag(W, nblk,      kb, K, lane);
        v16bf b1 = load_b_frag(W, nblk + 16, kb, K, lane);
#pragma unroll
        for (int i = 0; i < 4; ++i) {
            v16bf a = load_a_frag(A, mblk + i * 16 + arow, kb, K, lane);
            acc[i][0] = __builtin_amdgcn_wmma_f32_16x16x32_bf16(
                false, a, false, b0, (short)0, acc[i][0], false, false);
            acc[i][1] = __builtin_amdgcn_wmma_f32_16x16x32_bf16(
                false, a, false, b1, (short)0, acc[i][1], false, false);
        }
    }

    const int rofs = (lane >> 4) << 3;   // +0 (lanes 0-15) or +8 (lanes 16-31)
#pragma unroll
    for (int i = 0; i < 4; ++i) {
#pragma unroll
        for (int j = 0; j < 2; ++j) {
            int col = nblk + j * 16 + (lane & 15);
            float bcol = bias[col];
#pragma unroll
            for (int r = 0; r < 8; ++r) {
                int row = mblk + i * 16 + r + rofs;
                float v = (acc[i][j][r] + bcol) * alpha;
                if (GELU) v = gelu_exact(v);
                store_out(C, (size_t)row * N + col, v);
            }
        }
    }
}

// ---- small helpers ---------------------------------------------------------

__global__ void cvt_f32_bf16(const float* __restrict__ in,
                             __hip_bfloat16* __restrict__ out, long n) {
    long i = (long)blockIdx.x * blockDim.x + threadIdx.x;
    if (i < n) out[i] = __float2bfloat16(in[i]);
}

// txt[b,d] = text_features[b,:] . W_text[d,:] + b_text[d]
__global__ void txt_kernel(const float* __restrict__ tf,
                           const float* __restrict__ Wt,
                           const float* __restrict__ bt,
                           float* __restrict__ txt) {
    int d = blockIdx.x * blockDim.x + threadIdx.x;   // 0..1023
    int b = blockIdx.y;
    const float* t = tf + (size_t)b * TDq;
    const float* w = Wt + (size_t)d * TDq;
    float s = bt[d];
    for (int k = 0; k < TDq; ++k) s = fmaf(t[k], w[k], s);
    txt[b * Dq + d] = s;
}

// xq = bf16(x + txt broadcast over N)
__global__ void make_xq(const float* __restrict__ x, const float* __restrict__ txt,
                        __hip_bfloat16* __restrict__ xq, long n) {
    long i = (long)blockIdx.x * blockDim.x + threadIdx.x;
    if (i < n) {
        int d  = (int)(i & (Dq - 1));
        long b = i / ((long)Nq * Dq);
        xq[i] = __float2bfloat16(x[i] + txt[b * Dq + d]);
    }
}

// LayerNorm over D=1024; one block (256 threads) per row.
template <typename OutT>
__global__ __launch_bounds__(256)
void ln_kernel(const float* __restrict__ X, const float* __restrict__ g,
               const float* __restrict__ beta, OutT* __restrict__ Y) {
    const size_t row = blockIdx.x;
    const float* x = X + row * Dq;
    float s = 0.f, s2 = 0.f;
    for (int i = threadIdx.x; i < Dq; i += 256) {
        float v = x[i];
        s += v; s2 += v * v;
    }
    __shared__ float rs[256], rq[256];
    rs[threadIdx.x] = s; rq[threadIdx.x] = s2;
    __syncthreads();
    for (int st = 128; st > 0; st >>= 1) {
        if (threadIdx.x < st) {
            rs[threadIdx.x] += rs[threadIdx.x + st];
            rq[threadIdx.x] += rq[threadIdx.x + st];
        }
        __syncthreads();
    }
    float mean = rs[0] * (1.f / Dq);
    float var  = rq[0] * (1.f / Dq) - mean * mean;
    float rstd = rsqrtf(var + 1e-6f);
    for (int i = threadIdx.x; i < Dq; i += 256)
        store_out(Y, row * Dq + i, (x[i] - mean) * rstd * g[i] + beta[i]);
}

// Depthwise 4x4 stride-4 conv on q (B,24,24,D) -> qs_bf16 (B*36, D)
__global__ void dwconv_kernel(const float* __restrict__ q,
                              const float* __restrict__ cw,
                              const float* __restrict__ cb,
                              __hip_bfloat16* __restrict__ qs) {
    int d  = blockIdx.x * blockDim.x + threadIdx.x;
    int t  = blockIdx.y;
    int b  = blockIdx.z;
    int ty = t / 6, tx = t % 6;
    float s = cb[d];
#pragma unroll
    for (int i = 0; i < 4; ++i)
#pragma unroll
        for (int j = 0; j < 4; ++j) {
            int n = (ty * 4 + i) * Gq + tx * 4 + j;
            s = fmaf(q[((size_t)b * Nq + n) * Dq + d], cw[d * 16 + i * 4 + j], s);
        }
    qs[((size_t)b * Tq + t) * Dq + d] = __float2bfloat16(s);
}

// Windowed attention: per (bt, head): 1 query vs 16 keys/values of dim 128.
// One wave per head (8 waves/block), one block per bt. Lane owns 4 dims.
__global__ __launch_bounds__(256)
void attn_kernel(const float* __restrict__ qh, const float* __restrict__ kh,
                 const float* __restrict__ vh, __hip_bfloat16* __restrict__ o) {
    const int bt = blockIdx.x;            // 0..1151
    const int b  = bt / Tq, t = bt % Tq;
    const int ty = t / 6, tx = t % 6;
    const int h    = threadIdx.x >> 5;    // head 0..7
    const int lane = threadIdx.x & 31;

    const float* qp = qh + (size_t)bt * Dq + h * HDq + lane * 4;
    float q0 = qp[0], q1 = qp[1], q2 = qp[2], q3 = qp[3];

    float s[16];
#pragma unroll
    for (int e = 0; e < 16; ++e) {
        int i = e >> 2, j = e & 3;
        int n = (ty * 4 + i) * Gq + tx * 4 + j;
        const float* kp = kh + ((size_t)b * Nq + n) * Dq + h * HDq + lane * 4;
        float p = q0 * kp[0] + q1 * kp[1] + q2 * kp[2] + q3 * kp[3];
#pragma unroll
        for (int off = 16; off >= 1; off >>= 1) p += __shfl_xor(p, off, 32);
        s[e] = p;                          // all lanes hold the full dot
    }
    float m = s[0];
#pragma unroll
    for (int e = 1; e < 16; ++e) m = fmaxf(m, s[e]);
    float sum = 0.f;
#pragma unroll
    for (int e = 0; e < 16; ++e) { s[e] = expf(s[e] - m); sum += s[e]; }
    float inv = 1.f / sum;

    float o0 = 0.f, o1 = 0.f, o2 = 0.f, o3 = 0.f;
#pragma unroll
    for (int e = 0; e < 16; ++e) {
        int i = e >> 2, j = e & 3;
        int n = (ty * 4 + i) * Gq + tx * 4 + j;
        const float* vp = vh + ((size_t)b * Nq + n) * Dq + h * HDq + lane * 4;
        o0 = fmaf(s[e], vp[0], o0); o1 = fmaf(s[e], vp[1], o1);
        o2 = fmaf(s[e], vp[2], o2); o3 = fmaf(s[e], vp[3], o3);
    }
    __hip_bfloat16* op = o + (size_t)bt * Dq + h * HDq + lane * 4;
    op[0] = __float2bfloat16(o0 * inv);
    op[1] = __float2bfloat16(o1 * inv);
    op[2] = __float2bfloat16(o2 * inv);
    op[3] = __float2bfloat16(o3 * inv);
}

// ---------------------------------------------------------------------------

extern "C" void kernel_launch(void* const* d_in, const int* in_sizes, int n_in,
                              void* d_out, int out_size, void* d_ws, size_t ws_size,
                              hipStream_t stream) {
    const float* x_multi = (const float*)d_in[0];
    const float* text_f  = (const float*)d_in[1];
    const float* x       = (const float*)d_in[2];
    const float* W_text  = (const float*)d_in[3];
    const float* b_text  = (const float*)d_in[4];
    const float* q1_W    = (const float*)d_in[5];
    const float* q1_b    = (const float*)d_in[6];
    const float* q2_W    = (const float*)d_in[7];
    const float* q2_b    = (const float*)d_in[8];
    const float* q_g     = (const float*)d_in[9];
    const float* q_beta  = (const float*)d_in[10];
    const float* k1_W    = (const float*)d_in[11];
    const float* k1_b    = (const float*)d_in[12];
    const float* k2_W    = (const float*)d_in[13];
    const float* k2_b    = (const float*)d_in[14];
    const float* k_g     = (const float*)d_in[15];
    const float* k_beta  = (const float*)d_in[16];
    const float* v1_W    = (const float*)d_in[17];
    const float* v1_b    = (const float*)d_in[18];
    const float* v2_W    = (const float*)d_in[19];
    const float* v2_b    = (const float*)d_in[20];
    const float* v_g     = (const float*)d_in[21];
    const float* v_beta  = (const float*)d_in[22];
    const float* conv_W  = (const float*)d_in[23];
    const float* conv_b  = (const float*)d_in[24];
    const float* in_W    = (const float*)d_in[25];
    const float* in_b    = (const float*)d_in[26];
    const float* out_W   = (const float*)d_in[27];
    const float* out_b   = (const float*)d_in[28];
    float* out = (float*)d_out;

    // -------- workspace carve-up (256B aligned) --------
    uintptr_t base = (uintptr_t)d_ws;
    auto alloc = [&](size_t bytes) -> void* {
        uintptr_t r = (base + 255) & ~(uintptr_t)255;
        base = r + bytes;
        return (void*)r;
    };
    typedef __hip_bfloat16 bf;
    bf*    q1_bf   = (bf*)alloc((size_t)Dq * Dq * 2);
    bf*    q2_bf   = (bf*)alloc((size_t)Dq * Dq * 2);
    bf*    k1_bf   = (bf*)alloc((size_t)Dq * HSq * 2);
    bf*    k2_bf   = (bf*)alloc((size_t)Dq * Dq * 2);
    bf*    v1_bf   = (bf*)alloc((size_t)Dq * HSq * 2);
    bf*    v2_bf   = (bf*)alloc((size_t)Dq * Dq * 2);
    bf*    inW_bf  = (bf*)alloc((size_t)3 * Dq * Dq * 2);
    bf*    outW_bf = (bf*)alloc((size_t)Dq * Dq * 2);
    bf*    xm_bf   = (bf*)alloc((size_t)ROWS * HSq * 2);
    bf*    xq_bf   = (bf*)alloc((size_t)ROWS * Dq * 2);
    bf*    h_bf    = (bf*)alloc((size_t)ROWS * Dq * 2);
    float* y_f32   = (float*)alloc((size_t)ROWS * Dq * 4);
    float* txt_f32 = (float*)alloc((size_t)Bq * Dq * 4);
    float* qln_f32 = (float*)alloc((size_t)ROWS * Dq * 4);
    bf*    kln_bf  = (bf*)alloc((size_t)ROWS * Dq * 2);
    bf*    vln_bf  = (bf*)alloc((size_t)ROWS * Dq * 2);
    bf*    qs_bf   = (bf*)alloc((size_t)BTq * Dq * 2);
    float* qh_f32  = (float*)alloc((size_t)BTq * Dq * 4);
    float* kh_f32  = (float*)alloc((size_t)ROWS * Dq * 4);
    float* vh_f32  = (float*)alloc((size_t)ROWS * Dq * 4);
    bf*    o_bf    = (bf*)alloc((size_t)BTq * Dq * 2);
    (void)ws_size; (void)n_in; (void)in_sizes; (void)out_size;

    const dim3 blk(256);
    auto cvt = [&](const float* src, bf* dst, long n) {
        cvt_f32_bf16<<<dim3((unsigned)((n + 255) / 256)), blk, 0, stream>>>(src, dst, n);
    };

    // -------- stage 0: bf16 weight / activation conversion --------
    cvt(q1_W, q1_bf, (long)Dq * Dq);
    cvt(q2_W, q2_bf, (long)Dq * Dq);
    cvt(k1_W, k1_bf, (long)Dq * HSq);
    cvt(k2_W, k2_bf, (long)Dq * Dq);
    cvt(v1_W, v1_bf, (long)Dq * HSq);
    cvt(v2_W, v2_bf, (long)Dq * Dq);
    cvt(in_W, inW_bf, (long)3 * Dq * Dq);
    cvt(out_W, outW_bf, (long)Dq * Dq);
    cvt(x_multi, xm_bf, (long)ROWS * HSq);

    // -------- stage 1: txt projection + xq --------
    txt_kernel<<<dim3(Dq / 256, Bq), blk, 0, stream>>>(text_f, W_text, b_text, txt_f32);
    make_xq<<<dim3((unsigned)(((long)ROWS * Dq + 255) / 256)), blk, 0, stream>>>(
        x, txt_f32, xq_bf, (long)ROWS * Dq);

    const float one = 1.0f;
    const float scale = 1.0f / sqrtf((float)HDq);
    auto gemm_grid = [](int M, int N) { return dim3(N / 128, M / 128); };

    // -------- stage 2: q chain --------
    gemm_bt_wmma<bf, true ><<<gemm_grid(ROWS, Dq), blk, 0, stream>>>(
        xq_bf, q1_bf, q1_b, h_bf, ROWS, Dq, Dq, one);
    gemm_bt_wmma<float, false><<<gemm_grid(ROWS, Dq), blk, 0, stream>>>(
        h_bf, q2_bf, q2_b, y_f32, ROWS, Dq, Dq, one);
    ln_kernel<float><<<dim3(ROWS), blk, 0, stream>>>(y_f32, q_g, q_beta, qln_f32);

    // -------- stage 3: depthwise 4x4 stride-4 conv -> query windows --------
    dwconv_kernel<<<dim3(Dq / 256, Tq, Bq), blk, 0, stream>>>(qln_f32, conv_W, conv_b, qs_bf);

    // -------- stage 4: k chain --------
    gemm_bt_wmma<bf, true ><<<gemm_grid(ROWS, Dq), blk, 0, stream>>>(
        xm_bf, k1_bf, k1_b, h_bf, ROWS, Dq, HSq, one);
    gemm_bt_wmma<float, false><<<gemm_grid(ROWS, Dq), blk, 0, stream>>>(
        h_bf, k2_bf, k2_b, y_f32, ROWS, Dq, Dq, one);
    ln_kernel<bf><<<dim3(ROWS), blk, 0, stream>>>(y_f32, k_g, k_beta, kln_bf);

    // -------- stage 5: v chain --------
    gemm_bt_wmma<bf, true ><<<gemm_grid(ROWS, Dq), blk, 0, stream>>>(
        xm_bf, v1_bf, v1_b, h_bf, ROWS, Dq, HSq, one);
    gemm_bt_wmma<float, false><<<gemm_grid(ROWS, Dq), blk, 0, stream>>>(
        h_bf, v2_bf, v2_b, y_f32, ROWS, Dq, Dq, one);
    ln_kernel<bf><<<dim3(ROWS), blk, 0, stream>>>(y_f32, v_g, v_beta, vln_bf);

    // -------- stage 6: q/k/v head projections (in_W slices are contiguous) --
    gemm_bt_wmma<float, false><<<gemm_grid(BTq, Dq), blk, 0, stream>>>(
        qs_bf, inW_bf, in_b, qh_f32, BTq, Dq, Dq, scale);
    gemm_bt_wmma<float, false><<<gemm_grid(ROWS, Dq), blk, 0, stream>>>(
        kln_bf, inW_bf + (size_t)Dq * Dq, in_b + Dq, kh_f32, ROWS, Dq, Dq, one);
    gemm_bt_wmma<float, false><<<gemm_grid(ROWS, Dq), blk, 0, stream>>>(
        vln_bf, inW_bf + (size_t)2 * Dq * Dq, in_b + 2 * Dq, vh_f32, ROWS, Dq, Dq, one);

    // -------- stage 7: windowed softmax attention --------
    attn_kernel<<<dim3(BTq), blk, 0, stream>>>(qh_f32, kh_f32, vh_f32, o_bf);

    // -------- stage 8: output projection -> d_out (fp32) --------
    gemm_bt_wmma<float, false><<<gemm_grid(BTq, Dq), blk, 0, stream>>>(
        o_bf, outW_bf, out_b, out, BTq, Dq, Dq, one);
}
